// ParsingFuseAttention_33097017983146
// MI455X (gfx1250) — matrix operand order repaired and checked by reading, not compiled, not run on hardware
//
#include <hip/hip_runtime.h>
#include <hip/hip_bf16.h>

// ---------------------------------------------------------------------------
// ParsingFuseAttention for MI455X (gfx1250, wave32, WMMA).
// All GEMM-shaped work (7 convs as implicit GEMM, Q^T K, V*softmax x2) runs on
// v_wmma_f32_16x16x32_f16 (f16 in, f32 accumulate).
// E and E^T stored f16 (67MB each) -> both L2-resident (192MB) across reads.
// Conv K-dim is tap-major (tap*CinPad+ci, CinPad%32==0, inputs+weights zero
// padded): compile-time taps, no div/mod, no branches in the inner loop.
// Register blocking: conv reuses each 16-load activation gather across MT
// M-tiles; energy reuses the Q operand across 4 m-tiles.  AV is block-
// cooperative: 4 waves share one exp()'d E tile staged in LDS (double
// buffered), so each E element is exponentiated once per softmax direction.
// ---------------------------------------------------------------------------

typedef __attribute__((ext_vector_type(16))) _Float16 v16h;
typedef __attribute__((ext_vector_type(8)))  _Float16 v8h;
typedef __attribute__((ext_vector_type(4)))  _Float16 v4h;
typedef __attribute__((ext_vector_type(8)))  float    v8f;

#define BATCH 2
#define CDIM  256
#define NCLS  19
#define QKD   32
#define IMG_H 64
#define IMG_W 64
#define NPIX  4096   // 64*64

// ---------------------------------------------------------------------------
// f32 -> f16 channel-pack:  dst[b][dstOff+c][pix] = src[b][c][pix]
// ---------------------------------------------------------------------------
__global__ void pack_f16_kernel(const float* __restrict__ src, _Float16* __restrict__ dst,
                                int srcCB, int dstCB, int dstOff, int nch) {
  size_t i = (size_t)blockIdx.x * blockDim.x + threadIdx.x;
  size_t tot = (size_t)BATCH * nch * NPIX;
  if (i >= tot) return;
  int pix = (int)(i & (NPIX - 1));
  int c   = (int)((i >> 12) % nch);
  int b   = (int)((i >> 12) / nch);
  dst[((size_t)b * dstCB + dstOff + c) * NPIX + pix] =
      (_Float16)src[((size_t)b * srcCB + c) * NPIX + pix];
}

// Zero-fill the padding channels [Cin..CinPad) of an assembled input buffer.
__global__ void zero_pad_kernel(_Float16* __restrict__ buf, int CinPad, int Cin) {
  size_t i = (size_t)blockIdx.x * blockDim.x + threadIdx.x;
  int padc = CinPad - Cin;
  size_t tot = (size_t)BATCH * padc * NPIX;
  if (i >= tot) return;
  int pix = (int)(i & (NPIX - 1));
  int c   = (int)((i >> 12) % padc);
  int b   = (int)((i >> 12) / padc);
  buf[((size_t)b * CinPad + Cin + c) * NPIX + pix] = (_Float16)0.f;
}

// ---------------------------------------------------------------------------
// Weight pack, tap-major: dst[r][tap*CinPad+ci] = w[r][ci][tap], zero padded.
// dst is [Rows][9*CinPad] f16.
// ---------------------------------------------------------------------------
__global__ void wpack_kernel(const float* __restrict__ w, _Float16* __restrict__ dst,
                             int Cout, int Cin, int CinPad, int Rows) {
  size_t i = (size_t)blockIdx.x * blockDim.x + threadIdx.x;
  size_t Kpad = (size_t)9 * CinPad;
  if (i >= (size_t)Rows * Kpad) return;
  int k   = (int)(i % Kpad);
  int r   = (int)(i / Kpad);
  int tap = k / CinPad;
  int ci  = k - tap * CinPad;
  dst[i] = (r < Cout && ci < Cin) ? (_Float16)w[((size_t)r * Cin + ci) * 9 + tap]
                                  : (_Float16)0.f;
}

// ---------------------------------------------------------------------------
// Implicit-GEMM 3x3 SAME conv on WMMA, tap-major K, MT M-tiles per wave.
// in  : [B][CinPad][NPIX] f16 (pad channels zeroed)
// wgt : [Mtiles*16][9*CinPad] f16 (zero padded)
// One wave computes (MT*16 oc) x (16 pixels along W); the 16-load activation
// gather (B operand) is reused by MT WMMAs.
// post: 0 = f16 store, 1 = f16 store * mask, 2 = f32 store,
//       3 = f16 transposed store  out[b][pix][outCB] (for Q/K).
// ---------------------------------------------------------------------------
template <int MT>
__global__ __launch_bounds__(256) void wmma_conv_kernel(
    const _Float16* __restrict__ in, const _Float16* __restrict__ wgt,
    const float* __restrict__ bias, const float* __restrict__ mask,
    void* __restrict__ out,
    int CinPad, int Mtiles, int Cout, int outCB, int outOff, int post) {
  const int wv   = (blockIdx.x * blockDim.x + threadIdx.x) >> 5;
  const int lane = threadIdx.x & 31;
  const int l16  = lane & 15, half = lane >> 4;
  const int PixTiles = (BATCH * NPIX) / 16;           // 512
  const int mgroups  = Mtiles / MT;
  if (wv >= mgroups * PixTiles) return;               // wave-uniform (grids exact)
  const int mg = wv / PixTiles;
  const int pt = wv - mg * PixTiles;
  const int w0 = (pt & 3) << 4;                       // 4 tiles along W
  const int h  = (pt >> 2) & (IMG_H - 1);
  const int b  = pt >> 8;
  const int wpix = w0 + l16;
  const _Float16* inb = in + (size_t)b * CinPad * NPIX;
  const _Float16* wrow[MT];
#pragma unroll
  for (int mi = 0; mi < MT; ++mi)
    wrow[mi] = wgt + (size_t)((mg * MT + mi) * 16 + l16) * (9 * CinPad);

  v8f acc[MT] = {};
#pragma unroll
  for (int tap = 0; tap < 9; ++tap) {
    const int dh = tap / 3 - 1, dw = tap % 3 - 1;     // compile-time
    const int hh = h + dh;
    if ((unsigned)hh >= IMG_H) continue;              // wave-uniform skip
    const int ww = wpix + dw;
    const bool ok = (unsigned)ww < IMG_W;             // per-lane, per-tap const
    const _Float16* bbase = inb + hh * IMG_W + (ok ? ww : 0);
    for (int cb = 0; cb < CinPad; cb += 32) {
      // B: 32x16 activations; lane column = pixel, K rows = channels.
      v16h bv;
#pragma unroll
      for (int j = 0; j < 16; ++j) {
        _Float16 v = bbase[(size_t)(cb + half * 16 + j) * NPIX];
        bv[j] = ok ? v : (_Float16)0.f;
      }
      // A: MT weight tiles, contiguous K; each reuses bv.
#pragma unroll
      for (int mi = 0; mi < MT; ++mi) {
        const _Float16* wp = wrow[mi] + tap * CinPad + cb;
        v8h alo = *(const v8h*)(wp + half * 16);
        v8h ahi = *(const v8h*)(wp + half * 16 + 8);
        v16h a;
#pragma unroll
        for (int j = 0; j < 8; ++j) { a[j] = alo[j]; a[8 + j] = ahi[j]; }
        acc[mi] = __builtin_amdgcn_wmma_f32_16x16x32_f16(false, a, false, bv, (short)0,
                                                         acc[mi], false, false);
      }
    }
  }
  const int pix = h * IMG_W + w0 + l16;
#pragma unroll
  for (int mi = 0; mi < MT; ++mi) {
#pragma unroll
    for (int r = 0; r < 8; ++r) {
      int oc = (mg * MT + mi) * 16 + r + half * 8;
      if (oc >= Cout) continue;
      float val = acc[mi][r] + bias[oc];
      if (post == 1) val *= mask[((size_t)b * Cout + oc) * NPIX + pix];
      if (post == 2)
        ((float*)out)[((size_t)b * outCB + outOff + oc) * NPIX + pix] = val;
      else if (post == 3)
        ((_Float16*)out)[((size_t)b * NPIX + pix) * outCB + outOff + oc] = (_Float16)val;
      else
        ((_Float16*)out)[((size_t)b * outCB + outOff + oc) * NPIX + pix] = (_Float16)val;
    }
  }
}

// ---------------------------------------------------------------------------
// Energy: E[b][n][m] = sum_d qT[b][n][d] * kT[b][m][d].  K = 32 = QKD exactly.
// One wave: one n-tile x 4 m-tiles (Q operand registers reused 4x).
// Writes E and E^T (both f16, both L2-resident) so every later pass (stats +
// both AV modes) reads rows contiguously.
// ---------------------------------------------------------------------------
__global__ __launch_bounds__(256) void energy_kernel(
    const _Float16* __restrict__ qT, const _Float16* __restrict__ kT,
    _Float16* __restrict__ E, _Float16* __restrict__ Et) {
  const int wv   = (blockIdx.x * blockDim.x + threadIdx.x) >> 5;
  const int lane = threadIdx.x & 31;
  const int l16  = lane & 15, half = lane >> 4;
  int t = wv;
  const int mg = t & 63;  t >>= 6;                    // 4-tile m group
  const int n0 = (t & 255) << 4; t >>= 8;
  const int b  = t;
  const _Float16* qrow = qT + ((size_t)b * NPIX + n0 + l16) * QKD;
  v8h a0 = *(const v8h*)(qrow + half * 16);
  v8h a1 = *(const v8h*)(qrow + half * 16 + 8);
  v16h a;
#pragma unroll
  for (int j = 0; j < 8; ++j) { a[j] = a0[j]; a[8 + j] = a1[j]; }
  _Float16* Eb  = E  + (size_t)b * NPIX * NPIX;
  _Float16* Etb = Et + (size_t)b * NPIX * NPIX;
#pragma unroll
  for (int mi = 0; mi < 4; ++mi) {
    const int m0 = mg * 64 + mi * 16;
    const _Float16* krow = kT + ((size_t)b * NPIX + m0 + l16) * QKD;
    v8h b0 = *(const v8h*)(krow + half * 16);
    v8h b1 = *(const v8h*)(krow + half * 16 + 8);
    v16h bv;
#pragma unroll
    for (int j = 0; j < 8; ++j) { bv[j] = b0[j]; bv[8 + j] = b1[j]; }
    v8f c = {};
    c = __builtin_amdgcn_wmma_f32_16x16x32_f16(false, a, false, bv, (short)0, c, false, false);
#pragma unroll
    for (int r = 0; r < 8; ++r) {
      int n = n0 + r + half * 8;
      _Float16 v = (_Float16)c[r];
      Eb[(size_t)n * NPIX + m0 + l16] = v;
      Etb[(size_t)(m0 + l16) * NPIX + n] = v;
    }
  }
}

// ---------------------------------------------------------------------------
// Online softmax stats over rows of a [rows][NPIX] f16 matrix; one wave/row.
// Run on E for row-softmax stats, on E^T for column-softmax stats.
// ---------------------------------------------------------------------------
__global__ void row_stats_kernel(const _Float16* __restrict__ E,
                                 float* __restrict__ mx, float* __restrict__ sm) {
  int wv   = (blockIdx.x * blockDim.x + threadIdx.x) >> 5;   // b*NPIX + n
  int lane = threadIdx.x & 31;
  const _Float16* row = E + (size_t)wv * NPIX;
  float m = -1e30f, s = 0.f;
  for (int i = lane; i < NPIX; i += 32) {
    float v  = (float)row[i];
    float nm = fmaxf(m, v);
    s = s * __expf(m - nm) + __expf(v - nm);
    m = nm;
  }
  for (int off = 16; off; off >>= 1) {
    float om = __shfl_xor(m, off);
    float os = __shfl_xor(s, off);
    float nm = fmaxf(m, om);
    s = s * __expf(m - nm) + os * __expf(om - nm);
    m = nm;
  }
  if (lane == 0) { mx[wv] = m; sm[wv] = s; }
}

// ---------------------------------------------------------------------------
// AV GEMM, block-cooperative.  One 128-thread block (4 waves) per
// (mode, b, m-tile); wave w handles channel group w (64 channels, 4 c-tiles).
// mode 0: out_b[c,m]=sum_n V[c,n]*P[m,n]     (row softmax, rows of E)
// mode 1: out_c[c,m]=sum_n V[c,n]*Pt[n,m]    (col softmax, rows of E^T)
// Per 32-wide K chunk: 128 threads load the raw 16x32 E tile (b64 each),
// exp(x-max) once, stage f16 into double-buffered LDS; all 4 waves consume
// the shared tile (4 WMMAs each).  1/sum folded at store.
// ---------------------------------------------------------------------------
#define AV_PITCH 40   // 16x(32+8) f16 rows: 80B row pitch, 16B aligned, bank-spread
__global__ __launch_bounds__(128) void av_kernel(
    const _Float16* __restrict__ V, const _Float16* __restrict__ E,
    const _Float16* __restrict__ Et,
    const float* __restrict__ rmax, const float* __restrict__ rsum,
    const float* __restrict__ cmax, const float* __restrict__ csum,
    _Float16* __restrict__ cat) {
  __shared__ _Float16 etile[2][16][AV_PITCH];
  __shared__ float sbmax[16];
  const int tid  = threadIdx.x;
  const int wave = tid >> 5;
  const int lane = tid & 31;
  const int l16  = lane & 15, half = lane >> 4;
  const int blk  = blockIdx.x;                  // 2*B*256 = 1024 blocks
  const int m0   = (blk & 255) << 4;
  const int b    = (blk >> 8) & 1;
  const int mode = blk >> 9;
  const int m    = m0 + l16;
  const _Float16* Vb    = V + (size_t)b * CDIM * NPIX;
  const _Float16* Ebase = (mode ? Et : E) + (size_t)b * NPIX * NPIX;
  // staging role: thread covers tile row em (0..15), cols en..en+3
  const int em = tid >> 3;
  const int en = (tid & 7) << 2;
  const _Float16* erow = Ebase + (size_t)(m0 + em) * NPIX + en;
  if (tid < 16) sbmax[tid] = (mode ? cmax : rmax)[b * NPIX + m0 + tid];
  __syncthreads();
  const float bmax_s = sbmax[em];

  v8f acc[4] = {};
  for (int ib = 0; ib < NPIX; ib += 32) {
    const int buf = (ib >> 5) & 1;
    // produce: raw b64 load -> exp -> LDS (f16)
    v4h raw = *(const v4h*)(erow + ib);
    v4h ex;
#pragma unroll
    for (int jj = 0; jj < 4; ++jj) ex[jj] = (_Float16)__expf((float)raw[jj] - bmax_s);
    *(v4h*)&etile[buf][em][en] = ex;
    __syncthreads();
    // consume: shared exp'd B tile, 4 c-tiles per wave
    const _Float16* bp = &etile[buf][l16][half * 16];
    v8h e0 = *(const v8h*)bp;
    v8h e1 = *(const v8h*)(bp + 8);
    v16h bv;
#pragma unroll
    for (int j = 0; j < 8; ++j) { bv[j] = e0[j]; bv[8 + j] = e1[j]; }
#pragma unroll
    for (int ci = 0; ci < 4; ++ci) {
      const _Float16* vr = Vb + (size_t)(wave * 64 + ci * 16 + l16) * NPIX + ib + half * 16;
      v8h alo = *(const v8h*)vr;
      v8h ahi = *(const v8h*)(vr + 8);
      v16h a;
#pragma unroll
      for (int j = 0; j < 8; ++j) { a[j] = alo[j]; a[8 + j] = ahi[j]; }
      acc[ci] = __builtin_amdgcn_wmma_f32_16x16x32_f16(false, a, false, bv, (short)0,
                                                       acc[ci], false, false);
    }
  }
  const float inv = 1.f / (mode ? csum : rsum)[b * NPIX + m];
  _Float16* ob = cat + (size_t)b * 768 * NPIX + (size_t)(256 + (mode << 8)) * NPIX;
#pragma unroll
  for (int ci = 0; ci < 4; ++ci)
#pragma unroll
    for (int r = 0; r < 8; ++r)
      ob[(size_t)(wave * 64 + ci * 16 + r + half * 8) * NPIX + m] =
          (_Float16)(acc[ci][r] * inv);
}

// ---------------------------------------------------------------------------
// SE layer: mean over HW, 256->32->256 MLP, sigmoid, scale into cat[0:256].
// ---------------------------------------------------------------------------
__global__ void se_reduce_kernel(const float* __restrict__ comp, float* __restrict__ y) {
  __shared__ float red[256];
  int bc = blockIdx.x;
  const float* p = comp + (size_t)bc * NPIX;
  float s = 0.f;
  for (int i = threadIdx.x; i < NPIX; i += 256) s += p[i];
  red[threadIdx.x] = s;
  __syncthreads();
  for (int off = 128; off; off >>= 1) {
    if ((int)threadIdx.x < off) red[threadIdx.x] += red[threadIdx.x + off];
    __syncthreads();
  }
  if (threadIdx.x == 0) y[bc] = red[0] * (1.f / NPIX);
}

__global__ void se_mlp_kernel(const float* __restrict__ y, const float* __restrict__ w1,
                              const float* __restrict__ w2, float* __restrict__ s) {
  __shared__ float h[BATCH][32];
  int t = threadIdx.x;  // 512 threads
  if (t < BATCH * 32) {
    int b = t >> 5, j = t & 31;
    float acc = 0.f;
    for (int c = 0; c < CDIM; ++c) acc += y[b * CDIM + c] * w1[j * CDIM + c];
    h[b][j] = fmaxf(acc, 0.f);
  }
  __syncthreads();
  int b = t >> 8, c = t & 255;
  float acc = 0.f;
#pragma unroll
  for (int j = 0; j < 32; ++j) acc += h[b][j] * w2[c * 32 + j];
  s[t] = 1.f / (1.f + __expf(-acc));
}

__global__ void se_scale_kernel(const float* __restrict__ comp, const float* __restrict__ s,
                                _Float16* __restrict__ cat) {
  size_t i = (size_t)blockIdx.x * blockDim.x + threadIdx.x;  // B*256*NPIX
  int bc = (int)(i >> 12);
  int b = bc >> 8, c = bc & 255;
  cat[((size_t)b * 768 + c) * NPIX + (i & (NPIX - 1))] = (_Float16)(comp[i] * s[bc]);
}

// ---------------------------------------------------------------------------
extern "C" void kernel_launch(void* const* d_in, const int* in_sizes, int n_in,
                              void* d_out, int out_size, void* d_ws, size_t ws_size,
                              hipStream_t stream) {
  const float* x           = (const float*)d_in[0];
  const float* modal_mask  = (const float*)d_in[1];
  const float* amodal_mask = (const float*)d_in[2];
  const float* w_mp  = (const float*)d_in[3];
  const float* b_mp  = (const float*)d_in[4];
  const float* w_ap  = (const float*)d_in[5];
  const float* b_ap  = (const float*)d_in[6];
  const float* w_q   = (const float*)d_in[7];
  const float* b_q   = (const float*)d_in[8];
  const float* w_k   = (const float*)d_in[9];
  const float* b_k   = (const float*)d_in[10];
  const float* w_v   = (const float*)d_in[11];
  const float* b_v   = (const float*)d_in[12];
  const float* w_comp= (const float*)d_in[13];
  const float* b_comp= (const float*)d_in[14];
  const float* w_out = (const float*)d_in[15];
  const float* b_out = (const float*)d_in[16];
  const float* w_se1 = (const float*)d_in[17];
  const float* w_se2 = (const float*)d_in[18];
  float* dout = (float*)d_out;

  char* ws = (char*)d_ws;
  size_t off = 0;
  auto alloc = [&](size_t elems, size_t esize) -> void* {
    off = (off + 255) & ~(size_t)255;
    void* p = (void*)(ws + off);
    off += elems * esize;
    return p;
  };

  // Padded channel counts (multiples of 32)
  const int CP_X  = 256;   // x-only convs (mp/ap/v)
  const int CP_QK = 288;   // 275 -> 288
  const int CP_CP = 320;   // 294 -> 320
  const int CP_OT = 768;   // final conv

  _Float16* xin    = (_Float16*)alloc((size_t)BATCH * CP_X  * NPIX, 2);
  _Float16* qin    = (_Float16*)alloc((size_t)BATCH * CP_QK * NPIX, 2);
  _Float16* kin    = (_Float16*)alloc((size_t)BATCH * CP_QK * NPIX, 2);
  _Float16* compin = (_Float16*)alloc((size_t)BATCH * CP_CP * NPIX, 2);
  _Float16* qT     = (_Float16*)alloc((size_t)BATCH * NPIX * QKD, 2);
  _Float16* kT     = (_Float16*)alloc((size_t)BATCH * NPIX * QKD, 2);
  _Float16* v16    = (_Float16*)alloc((size_t)BATCH * CDIM * NPIX, 2);
  _Float16* E      = (_Float16*)alloc((size_t)BATCH * NPIX * NPIX, 2);
  _Float16* Et     = (_Float16*)alloc((size_t)BATCH * NPIX * NPIX, 2);
  _Float16* cat    = (_Float16*)alloc((size_t)BATCH * 3 * CDIM * NPIX, 2);
  float* comp = (float*)alloc((size_t)BATCH * CDIM * NPIX, 4);
  float* rmax = (float*)alloc(BATCH * NPIX, 4);
  float* rsum = (float*)alloc(BATCH * NPIX, 4);
  float* cmax = (float*)alloc(BATCH * NPIX, 4);
  float* csum = (float*)alloc(BATCH * NPIX, 4);
  float* yse  = (float*)alloc(BATCH * CDIM, 4);
  float* sse  = (float*)alloc(BATCH * CDIM, 4);
  _Float16* wmp16 = (_Float16*)alloc((size_t)32  * 9 * CP_X,  2);
  _Float16* wap16 = (_Float16*)alloc((size_t)32  * 9 * CP_X,  2);
  _Float16* wq16  = (_Float16*)alloc((size_t)32  * 9 * CP_QK, 2);
  _Float16* wk16  = (_Float16*)alloc((size_t)32  * 9 * CP_QK, 2);
  _Float16* wv16  = (_Float16*)alloc((size_t)256 * 9 * CP_X,  2);
  _Float16* wcp16 = (_Float16*)alloc((size_t)256 * 9 * CP_CP, 2);
  _Float16* wot16 = (_Float16*)alloc((size_t)256 * 9 * CP_OT, 2);
  (void)ws_size; (void)in_sizes; (void)n_in; (void)out_size;

  auto blk = [](size_t tot) { return (unsigned)((tot + 255) / 256); };

  // ---- input packs (f32 -> f16 channel concat) + channel padding ----
  pack_f16_kernel<<<blk((size_t)BATCH * CDIM * NPIX), 256, 0, stream>>>(x, xin, CDIM, CP_X, 0, CDIM);
  pack_f16_kernel<<<blk((size_t)BATCH * CDIM * NPIX), 256, 0, stream>>>(x, qin, CDIM, CP_QK, 0, CDIM);
  pack_f16_kernel<<<blk((size_t)BATCH * NCLS * NPIX), 256, 0, stream>>>(modal_mask, qin, NCLS, CP_QK, CDIM, NCLS);
  pack_f16_kernel<<<blk((size_t)BATCH * CDIM * NPIX), 256, 0, stream>>>(x, kin, CDIM, CP_QK, 0, CDIM);
  pack_f16_kernel<<<blk((size_t)BATCH * NCLS * NPIX), 256, 0, stream>>>(amodal_mask, kin, NCLS, CP_QK, CDIM, NCLS);
  pack_f16_kernel<<<blk((size_t)BATCH * CDIM * NPIX), 256, 0, stream>>>(x, compin, CDIM, CP_CP, 0, CDIM);
  zero_pad_kernel<<<blk((size_t)BATCH * (CP_QK - 275) * NPIX), 256, 0, stream>>>(qin, CP_QK, 275);
  zero_pad_kernel<<<blk((size_t)BATCH * (CP_QK - 275) * NPIX), 256, 0, stream>>>(kin, CP_QK, 275);
  zero_pad_kernel<<<blk((size_t)BATCH * (CP_CP - 294) * NPIX), 256, 0, stream>>>(compin, CP_CP, 294);

  // ---- weight packs (tap-major, zero padded) ----
  wpack_kernel<<<blk((size_t)32  * 9 * CP_X ), 256, 0, stream>>>(w_mp,   wmp16, NCLS, CDIM,     CP_X,  32);
  wpack_kernel<<<blk((size_t)32  * 9 * CP_X ), 256, 0, stream>>>(w_ap,   wap16, NCLS, CDIM,     CP_X,  32);
  wpack_kernel<<<blk((size_t)32  * 9 * CP_QK), 256, 0, stream>>>(w_q,    wq16,  QKD,  CDIM+NCLS, CP_QK, 32);
  wpack_kernel<<<blk((size_t)32  * 9 * CP_QK), 256, 0, stream>>>(w_k,    wk16,  QKD,  CDIM+NCLS, CP_QK, 32);
  wpack_kernel<<<blk((size_t)256 * 9 * CP_X ), 256, 0, stream>>>(w_v,    wv16,  CDIM, CDIM,     CP_X,  256);
  wpack_kernel<<<blk((size_t)256 * 9 * CP_CP), 256, 0, stream>>>(w_comp, wcp16, CDIM, CDIM+2*NCLS, CP_CP, 256);
  wpack_kernel<<<blk((size_t)256 * 9 * CP_OT), 256, 0, stream>>>(w_out,  wot16, CDIM, 3*CDIM,   CP_OT, 256);

  // ---- convs (WMMA implicit GEMM) ----
  // small convs: Mtiles=2, MT=2 -> 512 waves (64 blocks)
  wmma_conv_kernel<2><<<64, 256, 0, stream>>>(xin, wmp16, b_mp, modal_mask, compin,
      CP_X, 2, NCLS, CP_CP, CDIM, 1);
  wmma_conv_kernel<2><<<64, 256, 0, stream>>>(xin, wap16, b_ap, amodal_mask, compin,
      CP_X, 2, NCLS, CP_CP, CDIM + NCLS, 1);
  wmma_conv_kernel<2><<<64, 256, 0, stream>>>(qin, wq16, b_q, nullptr, qT,
      CP_QK, 2, QKD, QKD, 0, 3);
  wmma_conv_kernel<2><<<64, 256, 0, stream>>>(kin, wk16, b_k, nullptr, kT,
      CP_QK, 2, QKD, QKD, 0, 3);
  // big convs: Mtiles=16, MT=4 -> 2048 waves (256 blocks)
  wmma_conv_kernel<4><<<256, 256, 0, stream>>>(xin, wv16, b_v, nullptr, v16,
      CP_X, 16, CDIM, CDIM, 0, 0);
  wmma_conv_kernel<4><<<256, 256, 0, stream>>>(compin, wcp16, b_comp, nullptr, comp,
      CP_CP, 16, CDIM, CDIM, 0, 2);

  // ---- SE path -> cat channels [0..256) ----
  se_reduce_kernel<<<BATCH * CDIM, 256, 0, stream>>>(comp, yse);
  se_mlp_kernel<<<1, BATCH * CDIM, 0, stream>>>(yse, w_se1, w_se2, sse);
  se_scale_kernel<<<blk((size_t)BATCH * CDIM * NPIX), 256, 0, stream>>>(comp, sse, cat);

  // ---- attention: E = Q^T K (+E^T), stats, AV x2 -> cat channels [256..768) ----
  energy_kernel<<<BATCH * 256 * 64 / 8, 256, 0, stream>>>(qT, kT, E, Et);
  row_stats_kernel<<<BATCH * NPIX / 8, 256, 0, stream>>>(E,  rmax, rsum);
  row_stats_kernel<<<BATCH * NPIX / 8, 256, 0, stream>>>(Et, cmax, csum);
  av_kernel<<<2 * BATCH * 256, 128, 0, stream>>>(v16, E, Et, rmax, rsum,
                                                 cmax, csum, cat);

  // ---- final conv 768->256, f32 out + bias ----
  wmma_conv_kernel<4><<<256, 256, 0, stream>>>(cat, wot16, b_out, nullptr, dout,
      CP_OT, 16, CDIM, CDIM, 0, 2);
}
